// NBAGraphNet_21096879358554
// MI455X (gfx1250) — matrix operand order repaired and checked by reading, not compiled, not run on hardware
//
#include <hip/hip_runtime.h>
#include <hip/hip_bf16.h>

#define BDIM   256
#define NN     30
#define RN     5
#define NF     12
#define DN     32
#define KH     4
#define DHh    8
#define DHID   128
#define TABD   256
#define ADJ_ELEMS (RN * NN * NN)          // 4500 floats per game
#define ADJ_VEC4  (ADJ_ELEMS / 4)         // 1125 16-byte chunks

typedef __attribute__((ext_vector_type(16))) __bf16 bf16x16;
typedef __attribute__((ext_vector_type(8)))  __bf16 bf16x8;
typedef __attribute__((ext_vector_type(8)))  float  f32x8;

union AFrag {
  bf16x16 v;
  bf16x8  h2[2];
  __bf16  e[16];
};

struct NetParams {
  const float* x_tab; const float* node_feats; const float* adj;
  const int* home_idx; const int* away_idx;
  const float *npW, *npb, *npg, *npbe;
  const float *gW[2], *gb[2], *grw[2], *glg[2], *glb[2];
  const float *gatW, *gatAs, *gatAd, *glnG, *glnB;
  const float *tabW, *tabB, *tabG, *tabBe;
  const float *hW1,*hb1,*hg1,*hbe1,*hW2,*hb2,*hg2,*hbe2,*hW3,*hb3;
  const float *gateW, *gateB;
  const float *tW1,*tb1,*tW2,*tb2;
  float* out;
};

__device__ __forceinline__ float gelu_f(float x) {
  return 0.5f * x * (1.0f + erff(x * 0.70710678118654752f));
}

// A-fragment: 16x32 bf16, row-major LDS tile (stride DN), rows m0..m0+15
__device__ __forceinline__ bf16x16 load_afrag(const __bf16* tile, int m0, int lane) {
  const int half = lane >> 4;
  const int row  = (lane & 15) + m0;
  const int kb   = half << 4;                 // K base 0 or 16
  AFrag f;
  f.h2[0] = *(const bf16x8*)(tile + row * DN + kb);        // K = kb..kb+7
  f.h2[1] = *(const bf16x8*)(tile + row * DN + kb + 8);    // K = kb+8..kb+15
  return f.v;
}

// B-fragment: 32x16 bf16 from *transposed* LDS tile T[n][k] (stride DN), cols n0..n0+15
__device__ __forceinline__ bf16x16 load_bfrag(const __bf16* tileT, int n0, int lane) {
  const int col = lane & 15;
  const int kb  = (lane >> 4) << 4;           // K base 0 or 16
  return *(const bf16x16*)(tileT + (n0 + col) * DN + kb);
}

__device__ __forceinline__ void ln32(float* t, const float* g, const float* be) {
  float mu = 0.f;
  #pragma unroll
  for (int n = 0; n < DN; ++n) mu += t[n];
  mu *= (1.0f / DN);
  float var = 0.f;
  #pragma unroll
  for (int n = 0; n < DN; ++n) { float d = t[n] - mu; var += d * d; }
  var *= (1.0f / DN);
  const float inv = rsqrtf(var + 1e-5f);
  #pragma unroll
  for (int n = 0; n < DN; ++n) t[n] = (t[n] - mu) * inv * g[n] + be[n];
}

__device__ __forceinline__ void stats_serial(const float* x, int n, float* st) {
  float mu = 0.f;
  for (int i = 0; i < n; ++i) mu += x[i];
  mu /= (float)n;
  float var = 0.f;
  for (int i = 0; i < n; ++i) { float d = x[i] - mu; var += d * d; }
  var /= (float)n;
  st[0] = mu;
  st[1] = rsqrtf(var + 1e-5f);
}

__device__ __forceinline__ void wait_asynccnt0() {
#if defined(__has_builtin) && __has_builtin(__builtin_amdgcn_s_wait_asynccnt)
  __builtin_amdgcn_s_wait_asynccnt(0);
#else
  asm volatile("s_wait_asynccnt 0x0" ::: "memory");
#endif
}

__global__ __launch_bounds__(BDIM) void nba_graphnet_kernel(NetParams P) {
  const int b    = blockIdx.x;
  const int tid  = threadIdx.x;
  const int lane = tid & 31;
  const int wv   = __builtin_amdgcn_readfirstlane(tid >> 5);  // wave-uniform SGPR
  const int colX = lane & 15;
  const int hlfX = lane >> 4;

  __shared__ __align__(16) float  s_h[DN * DN];          // node features f32 (rows 30,31 = 0)
  __shared__ __align__(32) __bf16 s_hb[DN * DN];         // bf16 copy (A operand)
  __shared__ __align__(32) __bf16 s_Ab[RN * DN * DN];    // normalized adjacency, bf16, row-major
  // pool: first used as f32 adjacency staging (18000B), later as s_WT + s_supT (10240B each)
  __shared__ __align__(32) unsigned char s_pool[2 * RN * DN * DN * 2];
  __shared__ __align__(16) float  s_new[DN * DN];        // h_new / h_gat staging
  __shared__ __align__(16) float  s_Wh[DN * DN];         // GAT Wh row-major f32
  __shared__ __align__(32) __bf16 s_WhT[DN * DN];        // GAT Wh transposed bf16
  __shared__ __align__(32) __bf16 s_alpha[KH * DN * DN]; // attention, bf16, row-major
  __shared__ float s_dinv[RN * DN];
  __shared__ float s_relw[8];
  __shared__ float s_bsum[DN];
  __shared__ float s_src[KH * DN];
  __shared__ float s_dst[KH * DN];
  __shared__ float s_fused[2 * DHID];
  __shared__ float s_tab[DHID];
  __shared__ float s_z1[DHID];
  __shared__ float s_z2[64];
  __shared__ float s_t1[64];
  __shared__ float s_raw[DHID];
  __shared__ float s_stat[2];

  float*  s_adj  = (float*)s_pool;                        // 4500 f32 staging
  __bf16* s_WT   = (__bf16*)s_pool;                       // 5*32*32 bf16
  __bf16* s_supT = (__bf16*)(s_pool + RN * DN * DN * 2);  // 5*32*32 bf16

  // ---------- Stage 0: async DMA of this game's adjacency block into LDS ----------
  {
    const unsigned lds_base = (unsigned)(uintptr_t)s_adj;       // low 32 bits = LDS offset
    const unsigned long long gbase =
        (unsigned long long)(uintptr_t)(P.adj + (size_t)b * ADJ_ELEMS);
    for (int j = tid; j < ADJ_VEC4; j += BDIM) {
      const unsigned           ldso = lds_base + (unsigned)(j * 16);
      const unsigned long long ga   = gbase + (unsigned long long)(j * 16);
      asm volatile("global_load_async_to_lds_b128 %0, %1, off"
                   :: "v"(ldso), "v"(ga) : "memory");
    }
  }

  // ---------- overlap with DMA: zero pads + node projection ----------
  for (int j = tid; j < RN * DN * DN; j += BDIM) s_Ab[j] = (__bf16)0.0f;
  if (tid < 2 * DN) s_h[NN * DN + tid] = 0.0f;  // zero pad rows 30,31

  if (tid >= 192 && tid < 192 + NN) {
    const int m = tid - 192;
    const float* nf = P.node_feats + ((size_t)b * NN + m) * NF;
    float t[DN];
    #pragma unroll
    for (int n = 0; n < DN; ++n) {
      float a = P.npb[n];
      #pragma unroll
      for (int f = 0; f < NF; ++f) a += nf[f] * P.npW[f * DN + n];
      t[n] = a;
    }
    ln32(t, P.npg, P.npbe);
    #pragma unroll
    for (int n = 0; n < DN; ++n) s_h[m * DN + n] = gelu_f(t[n]);
  }

  wait_asynccnt0();
  __syncthreads();

  // ---------- Stage 1: adjacency row-sums (from LDS) ----------
  if (tid < RN * NN) {
    const int r = tid / NN, m = tid % NN;
    const float* arow = s_adj + (r * NN + m) * NN;
    float s = 1.0f;  // self loop
    for (int n = 0; n < NN; ++n) s += arow[n];
    s_dinv[r * DN + m] = rsqrtf(fmaxf(s, 1e-8f));
  }
  __syncthreads();

  // ---------- Stage 2: normalized adjacency D^-1/2 (A+I) D^-1/2 -> bf16 ----------
  if (tid < RN * NN) {
    const int r = tid / NN, m = tid % NN;
    const float* arow = s_adj + (r * NN + m) * NN;
    const float dm = s_dinv[r * DN + m];
    for (int n = 0; n < NN; ++n) {
      const float av = arow[n] + ((n == m) ? 1.0f : 0.0f);
      s_Ab[(r * DN + m) * DN + n] = (__bf16)(dm * av * s_dinv[r * DN + n]);
    }
  }
  __syncthreads();

  // ---------- GCN layers ----------
  for (int layer = 0; layer < 2; ++layer) {
    const float* Wl  = P.gW[layer];
    const float* bl  = P.gb[layer];
    const float* rwl = P.grw[layer];

    for (int j = tid; j < RN * DN * DN; j += BDIM) {
      const int r = j >> 10, rem = j & 1023, n = rem >> 5, k = rem & 31;
      s_WT[j] = (__bf16)(Wl[r * 1024 + k * DN + n]);   // transposed store
    }
    for (int j = tid; j < DN * DN; j += BDIM) s_hb[j] = (__bf16)(s_h[j]);
    if (tid == 0) {
      float mx = rwl[0];
      for (int r = 1; r < RN; ++r) mx = fmaxf(mx, rwl[r]);
      float s = 0.f, ex[RN];
      for (int r = 0; r < RN; ++r) { ex[r] = expf(rwl[r] - mx); s += ex[r]; }
      for (int r = 0; r < RN; ++r) s_relw[r] = ex[r] / s;
    }
    __syncthreads();

    if (tid < DN) {
      float a = 0.f;
      for (int r = 0; r < RN; ++r) a += s_relw[r] * bl[r * DN + tid];
      s_bsum[tid] = a;
    }

    // support_r = (h @ W_r) * softmax(rel_w)[r]  -> s_supT (bf16, transposed)
    for (int job = wv; job < RN * 4; job += 8) {
      const int r = job >> 2, t = job & 3;
      const int m0 = (t >> 1) << 4, n0 = (t & 1) << 4;
      bf16x16 a  = load_afrag(s_hb, m0, lane);
      bf16x16 bm = load_bfrag(s_WT + r * 1024, n0, lane);
      f32x8 acc;
      #pragma unroll
      for (int v = 0; v < 8; ++v) acc[v] = 0.f;
      acc = __builtin_amdgcn_wmma_f32_16x16x32_bf16(false, a, false, bm,
                                                    (short)0, acc, false, false);
      const float wr = s_relw[r];
      #pragma unroll
      for (int v = 0; v < 8; ++v)
        s_supT[r * 1024 + (n0 + colX) * DN + m0 + hlfX * 8 + v] = (__bf16)(acc[v] * wr);
    }
    __syncthreads();

    // h_new = sum_r A_r @ support_r + weighted bias  (5 chained WMMAs per tile)
    if (wv < 4) {
      const int m0 = (wv >> 1) << 4, n0 = (wv & 1) << 4;
      f32x8 acc;
      #pragma unroll
      for (int v = 0; v < 8; ++v) acc[v] = 0.f;
      #pragma unroll
      for (int r = 0; r < RN; ++r) {
        bf16x16 a  = load_afrag(s_Ab + r * 1024, m0, lane);
        bf16x16 bm = load_bfrag(s_supT + r * 1024, n0, lane);
        acc = __builtin_amdgcn_wmma_f32_16x16x32_bf16(false, a, false, bm,
                                                      (short)0, acc, false, false);
      }
      const float bs = s_bsum[n0 + colX];
      #pragma unroll
      for (int v = 0; v < 8; ++v)
        s_new[(m0 + hlfX * 8 + v) * DN + n0 + colX] = acc[v] + bs;
    }
    __syncthreads();

    if (tid < NN) {
      float t[DN];
      #pragma unroll
      for (int n = 0; n < DN; ++n)
        t[n] = s_h[tid * DN + n] + gelu_f(s_new[tid * DN + n]);
      ln32(t, P.glg[layer], P.glb[layer]);
      #pragma unroll
      for (int n = 0; n < DN; ++n) s_h[tid * DN + n] = t[n];
    }
    __syncthreads();
  }

  // ---------- GAT ----------
  for (int j = tid; j < DN * DN; j += BDIM) {
    const int nout = j >> 5, k = j & 31;
    const int head = nout >> 3, d = nout & 7;
    s_WT[j] = (__bf16)(P.gatW[(head * DN + k) * DHh + d]);  // concat-head W, transposed
    s_hb[j] = (__bf16)(s_h[j]);
  }
  __syncthreads();

  if (wv < 4) {  // Wh = h @ Wcat : 4 tiles
    const int m0 = (wv >> 1) << 4, n0 = (wv & 1) << 4;
    bf16x16 a  = load_afrag(s_hb, m0, lane);
    bf16x16 bm = load_bfrag(s_WT, n0, lane);
    f32x8 acc;
    #pragma unroll
    for (int v = 0; v < 8; ++v) acc[v] = 0.f;
    acc = __builtin_amdgcn_wmma_f32_16x16x32_bf16(false, a, false, bm,
                                                  (short)0, acc, false, false);
    #pragma unroll
    for (int v = 0; v < 8; ++v) {
      const int row = m0 + hlfX * 8 + v, nn_ = n0 + colX;
      s_Wh[row * DN + nn_]  = acc[v];
      s_WhT[nn_ * DN + row] = (__bf16)acc[v];
    }
  }
  __syncthreads();

  if (tid < KH * DN) {  // per-head src/dst scores
    const int k = tid >> 5, m = tid & 31;
    float sv = 0.f, dv = 0.f;
    if (m < NN) {
      #pragma unroll
      for (int d = 0; d < DHh; ++d) {
        const float wh = s_Wh[m * DN + k * DHh + d];
        sv += wh * P.gatAs[k * DHh + d];
        dv += wh * P.gatAd[k * DHh + d];
      }
    }
    s_src[tid] = sv; s_dst[tid] = dv;
  }
  __syncthreads();

  if (tid < KH * DN) {  // masked row softmax -> bf16 alpha
    const int k = tid >> 5, m = tid & 31;
    __bf16* arow = s_alpha + (k * DN + m) * DN;
    if (m < NN) {
      const float sm = s_src[k * DN + m];
      float ev[NN];
      float mx = -1e30f;
      #pragma unroll
      for (int n = 0; n < NN; ++n) {
        bool msk = (m == n);
        #pragma unroll
        for (int r = 0; r < RN; ++r)
          msk = msk || ((float)s_Ab[(r * DN + m) * DN + n] != 0.0f);
        float e = sm + s_dst[k * DN + n];
        e = (e > 0.f) ? e : 0.2f * e;   // LeakyReLU(0.2)
        ev[n] = msk ? e : -1e30f;
        if (msk) mx = fmaxf(mx, e);
      }
      float ssum = 0.f;
      #pragma unroll
      for (int n = 0; n < NN; ++n) {
        const float x = (ev[n] > -1e29f) ? expf(ev[n] - mx) : 0.0f;
        ev[n] = x; ssum += x;
      }
      const float inv = 1.0f / (ssum + 1e-10f);
      #pragma unroll
      for (int n = 0; n < DN; ++n)
        arow[n] = (__bf16)((n < NN) ? ev[n] * inv : 0.0f);
    } else {
      #pragma unroll
      for (int n = 0; n < DN; ++n) arow[n] = (__bf16)0.0f;
    }
  }
  __syncthreads();

  {  // h_gat[head] = alpha_head @ Wh_head : one WMMA per wave (8 jobs)
    const int head = wv >> 1, m0 = (wv & 1) << 4;
    bf16x16 a = load_afrag(s_alpha + head * 1024, m0, lane);
    AFrag fb;
    #pragma unroll
    for (int e2 = 0; e2 < 16; ++e2) fb.e[e2] = (__bf16)0.0f;
    const int kb = hlfX << 4;
    if (colX < DHh)
      fb.v = *(const bf16x16*)(s_WhT + (head * DHh + colX) * DN + kb);
    f32x8 acc;
    #pragma unroll
    for (int v = 0; v < 8; ++v) acc[v] = 0.f;
    acc = __builtin_amdgcn_wmma_f32_16x16x32_bf16(false, a, false, fb.v,
                                                  (short)0, acc, false, false);
    if (colX < DHh) {
      #pragma unroll
      for (int v = 0; v < 8; ++v)
        s_new[(m0 + hlfX * 8 + v) * DN + head * DHh + colX] = acc[v];
    }
  }
  __syncthreads();

  if (tid < NN) {
    float t[DN];
    #pragma unroll
    for (int n = 0; n < DN; ++n)
      t[n] = s_h[tid * DN + n] + gelu_f(s_new[tid * DN + n]);
    ln32(t, P.glnG, P.glnB);
    #pragma unroll
    for (int n = 0; n < DN; ++n) s_h[tid * DN + n] = t[n];
  }
  __syncthreads();

  // ---------- Heads ----------
  const int hm = P.home_idx[b], aw = P.away_idx[b];
  if (tid < DHID) {
    const int seg = tid >> 5, n = tid & 31;
    const float hh = s_h[hm * DN + n], ha = s_h[aw * DN + n];
    float v;
    if (seg == 0) v = hh;
    else if (seg == 1) v = ha;
    else if (seg == 2) v = hh - ha;
    else v = hh * ha;
    s_fused[tid] = v;

    float acc = P.tabB[tid];
    const float* xt = P.x_tab + (size_t)b * TABD;
    for (int i = 0; i < TABD; ++i) acc += xt[i] * P.tabW[i * DHID + tid];
    s_raw[tid] = acc;
  }
  __syncthreads();
  if (tid == 0) stats_serial(s_raw, DHID, s_stat);
  __syncthreads();
  if (tid < DHID) {
    const float x = (s_raw[tid] - s_stat[0]) * s_stat[1] * P.tabG[tid] + P.tabBe[tid];
    const float t = gelu_f(x);
    s_tab[tid] = t;
    s_fused[DHID + tid] = t;
  }
  __syncthreads();
  if (tid < DHID) {
    float acc = P.hb1[tid];
    for (int i = 0; i < 2 * DHID; ++i) acc += s_fused[i] * P.hW1[i * DHID + tid];
    s_raw[tid] = acc;
  }
  __syncthreads();
  if (tid == 0) stats_serial(s_raw, DHID, s_stat);
  __syncthreads();
  if (tid < DHID)
    s_z1[tid] = gelu_f((s_raw[tid] - s_stat[0]) * s_stat[1] * P.hg1[tid] + P.hbe1[tid]);
  __syncthreads();
  if (tid < 64) {
    float acc = P.hb2[tid];
    for (int i = 0; i < DHID; ++i) acc += s_z1[i] * P.hW2[i * 64 + tid];
    s_raw[tid] = acc;
  } else if (tid < 128) {
    const int j = tid - 64;
    float acc = P.tb1[j];
    for (int i = 0; i < DHID; ++i) acc += s_tab[i] * P.tW1[i * 64 + j];
    s_t1[j] = gelu_f(acc);
  }
  __syncthreads();
  if (tid == 0) stats_serial(s_raw, 64, s_stat);
  __syncthreads();
  if (tid < 64)
    s_z2[tid] = gelu_f((s_raw[tid] - s_stat[0]) * s_stat[1] * P.hg2[tid] + P.hbe2[tid]);
  __syncthreads();
  if (tid == 0) {
    float gl = P.hb3[0];
    for (int j = 0; j < 64; ++j) gl += s_z2[j] * P.hW3[j];
    float gt = P.gateB[0];
    for (int i = 0; i < 2 * DHID; ++i) gt += s_fused[i] * P.gateW[i];
    const float gate = 1.0f / (1.0f + expf(-gt));
    float tl = P.tb2[0];
    for (int j = 0; j < 64; ++j) tl += s_t1[j] * P.tW2[j];
    P.out[b] = gate * gl + (1.0f - gate) * tl;
  }
}

extern "C" void kernel_launch(void* const* d_in, const int* in_sizes, int n_in,
                              void* d_out, int out_size, void* d_ws, size_t ws_size,
                              hipStream_t stream) {
  NetParams P;
  P.x_tab      = (const float*)d_in[0];
  P.node_feats = (const float*)d_in[1];
  P.adj        = (const float*)d_in[2];
  P.home_idx   = (const int*)d_in[3];
  P.away_idx   = (const int*)d_in[4];
  P.npW  = (const float*)d_in[5];  P.npb  = (const float*)d_in[6];
  P.npg  = (const float*)d_in[7];  P.npbe = (const float*)d_in[8];
  for (int l = 0; l < 2; ++l) {
    P.gW[l]  = (const float*)d_in[9 + 5 * l];
    P.gb[l]  = (const float*)d_in[10 + 5 * l];
    P.grw[l] = (const float*)d_in[11 + 5 * l];
    P.glg[l] = (const float*)d_in[12 + 5 * l];
    P.glb[l] = (const float*)d_in[13 + 5 * l];
  }
  P.gatW  = (const float*)d_in[19]; P.gatAs = (const float*)d_in[20];
  P.gatAd = (const float*)d_in[21];
  P.glnG  = (const float*)d_in[22]; P.glnB  = (const float*)d_in[23];
  P.tabW  = (const float*)d_in[24]; P.tabB  = (const float*)d_in[25];
  P.tabG  = (const float*)d_in[26]; P.tabBe = (const float*)d_in[27];
  P.hW1 = (const float*)d_in[28]; P.hb1 = (const float*)d_in[29];
  P.hg1 = (const float*)d_in[30]; P.hbe1 = (const float*)d_in[31];
  P.hW2 = (const float*)d_in[32]; P.hb2 = (const float*)d_in[33];
  P.hg2 = (const float*)d_in[34]; P.hbe2 = (const float*)d_in[35];
  P.hW3 = (const float*)d_in[36]; P.hb3 = (const float*)d_in[37];
  P.gateW = (const float*)d_in[38]; P.gateB = (const float*)d_in[39];
  P.tW1 = (const float*)d_in[40]; P.tb1 = (const float*)d_in[41];
  P.tW2 = (const float*)d_in[42]; P.tb2 = (const float*)d_in[43];
  P.out = (float*)d_out;

  const int B = in_sizes[0] / TABD;
  hipLaunchKernelGGL(nba_graphnet_kernel, dim3(B), dim3(BDIM), 0, stream, P);
}
